// Block_40175124087290
// MI455X (gfx1250) — matrix operand order repaired and checked by reading, not compiled, not run on hardware
//
#include <hip/hip_runtime.h>
#include <hip/hip_bf16.h>
#include <stdint.h>

#define HWDIM 512
#define NB    4
#define NCIN  55
#define NRG   30
#define NCG   20

// tile geometry for the conv kernel
#define TW    32          // output tile width
#define THT   8           // output tile height
#define HALO  6           // max dilation*2 = 3*2
#define IW    (TW + 2*HALO)   // 44
#define IH    (THT + 2*HALO)  // 20
#define XT_HALFS (IH * IW * 64)   // 56320 bf16 in LDS x-tile
#define FT_HALFS (THT * TW * 32)  // 8192  bf16 in LDS feat-tile

typedef __attribute__((ext_vector_type(16))) __bf16        v16bf;
typedef __attribute__((ext_vector_type(8)))  float          v8f;
typedef __attribute__((ext_vector_type(8)))  unsigned short v8u;
typedef __attribute__((ext_vector_type(16))) unsigned short v16u;

__device__ __forceinline__ unsigned short f2bf(float f) {
  unsigned u = __float_as_uint(f);
  u += 0x7FFFu + ((u >> 16) & 1u);     // round-to-nearest-even
  return (unsigned short)(u >> 16);
}

__device__ __forceinline__ v16bf cat16(v8u lo, v8u hi) {
  v16u r = __builtin_shufflevector(lo, hi, 0,1,2,3,4,5,6,7,8,9,10,11,12,13,14,15);
  return __builtin_bit_cast(v16bf, r);
}

// ---------------- workspace layout (bytes) ----------------
static constexpr size_t OFF_XC  = 0;                                   // x cast to bf16, [B][H][W][64]
static constexpr size_t SZ_XC   = (size_t)NB * HWDIM * HWDIM * 64 * 2; // 128 MB
static constexpr size_t OFF_PW  = OFF_XC + SZ_XC;                      // conv B-fragments: 150 frags x 32 lanes x 16 halfs
static constexpr size_t SZ_PW   = (size_t)150 * 32 * 16 * 2;
static constexpr size_t OFF_PWT = OFF_PW + SZ_PW;                      // top 1x1 B-fragments: 3 x 32 x 16
static constexpr size_t SZ_PWT  = (size_t)3 * 32 * 16 * 2;
static constexpr size_t OFF_PWB = OFF_PWT + SZ_PWT;                    // bot 1x1 B-fragment: 32 x 16
static constexpr size_t SZ_PWB  = (size_t)32 * 16 * 2;
static constexpr size_t OFF_TS  = OFF_PWB + SZ_PWB;                    // top cell sums [600] f32
static constexpr size_t OFF_BS  = OFF_TS + 2560;                       // bot cell sums [600] f32
static constexpr size_t OFF_TM  = OFF_BS + 2560;                       // top cell means [600] f32
static constexpr size_t OFF_SG  = OFF_TM + 2560;                       // sigmoid(bot means) [600] f32

// ---------------- kernels ----------------

__global__ void zero_accum(float* ts, float* bs) {
  int t = threadIdx.x;
  if (t < 600) { ts[t] = 0.f; bs[t] = 0.f; }
}

// x [B][55][H][W] f32  ->  xc [B][H][W][64] bf16 (cin 55..63 zero)
__global__ __launch_bounds__(256) void cast_x(const float* __restrict__ x,
                                              unsigned short* __restrict__ xc) {
  int idx = blockIdx.x * 256 + threadIdx.x;       // 0 .. B*H*W-1
  int xw = idx & (HWDIM - 1);
  int t  = idx >> 9;
  int y  = t & (HWDIM - 1);
  int b  = t >> 9;
  unsigned short buf[64];
#pragma unroll
  for (int c = 0; c < 64; ++c) {
    float v = 0.f;
    if (c < NCIN) v = x[(((size_t)b * NCIN + c) * HWDIM + y) * HWDIM + xw];
    buf[c] = f2bf(v);
  }
  v8u* dst = (v8u*)(xc + (size_t)idx * 64);
  const v8u* src = (const v8u*)buf;
#pragma unroll
  for (int i = 0; i < 8; ++i) dst[i] = src[i];
}

// pack conv weights into 32x16 bf16 WMMA B-fragments in this file's contiguous k order (as A loads):
// element e of lane L:  K = (L>=16 ? 16 : 0) + e,  N = L & 15
__global__ void pack_conv_w(const float* w1, const float* w2, const float* w3,
                            unsigned short* out) {
  int f = blockIdx.x;           // 0..149 : ((conv*25 + tap)*2 + kc)
  int lane = threadIdx.x;
  int kc = f & 1;
  int tap = (f >> 1) % 25;
  int c = f / 50;
  const float* w = (c == 0) ? w1 : (c == 1) ? w2 : w3;
  int N  = lane & 15;
  int kh = (lane >= 16) ? 16 : 0;
  unsigned short* dst = out + (size_t)f * 512 + lane * 16;
  for (int e = 0; e < 16; ++e) {
    int cin = kc * 32 + kh + e;
    float v = 0.f;
    if (N < 6 && cin < NCIN) v = w[(N * NCIN + cin) * 25 + tap];
    dst[e] = f2bf(v);
  }
}

__global__ void pack_wt(const float* wt, unsigned short* out) {
  int nt = blockIdx.x;          // 0..2
  int lane = threadIdx.x;
  int N  = lane & 15;
  int kh = (lane >= 16) ? 16 : 0;
  int oc = nt * 16 + N;
  unsigned short* dst = out + (size_t)nt * 512 + lane * 16;
  for (int e = 0; e < 16; ++e) {
    int k = kh + e;
    float v = 0.f;
    if (oc < 36 && k < 18) v = wt[oc * 18 + k];
    dst[e] = f2bf(v);
  }
}

__global__ void pack_wb(const float* wb, unsigned short* out) {
  int lane = threadIdx.x;
  int N  = lane & 15;
  int kh = (lane >= 16) ? 16 : 0;
  unsigned short* dst = out + lane * 16;
  for (int e = 0; e < 16; ++e) {
    int k = kh + e;
    float v = 0.f;
    if (N == 0 && k < 18) v = wb[k];
    dst[e] = f2bf(v);
  }
}

// main fused kernel: 3 dilated 5x5 convs (implicit GEMM via WMMA) + ReLU,
// feat -> out channels 36..53, then 1x1 top/bot WMMAs + per-cell atomics.
__global__ __launch_bounds__(256) void conv_main(
    const unsigned short* __restrict__ xc,
    const unsigned short* __restrict__ pw,
    const unsigned short* __restrict__ pwt,
    const unsigned short* __restrict__ pwb,
    const float* __restrict__ b1, const float* __restrict__ b2,
    const float* __restrict__ b3, const float* __restrict__ bt,
    const float* __restrict__ bbv,
    const int* __restrict__ rowseg, const int* __restrict__ colseg,
    float* __restrict__ out, float* __restrict__ topsum, float* __restrict__ botsum)
{
  extern __shared__ unsigned short smem[];
  unsigned short* xt = smem;               // [IH*IW][64]
  unsigned short* ft = smem + XT_HALFS;    // [THT*TW][32]

  const int tid = threadIdx.x;
  const int bx = blockIdx.x;               // W / 32
  const int by = blockIdx.y;               // H / 8
  const int bi = blockIdx.z;               // batch

  // phase 0: zero LDS (covers OOB halo + padded channels)
  {
    v8u z = {};
    for (int i = tid; i < (XT_HALFS + FT_HALFS) / 8; i += 256)
      ((v8u*)smem)[i] = z;
  }
  __syncthreads();

  // phase 0b: async per-lane gather of the NHWC x-tile into LDS ([pixel][64] bf16)
  const int gx0 = bx * TW - HALO;
  const int gy0 = by * THT - HALO;
  for (int t = tid; t < IH * IW * 8; t += 256) {
    int part = t & 7;                      // 8 halfs per transfer
    int pix  = t >> 3;
    int r = pix / IW, cc = pix % IW;
    int gy = gy0 + r, gx = gx0 + cc;
    if ((unsigned)gy < HWDIM && (unsigned)gx < HWDIM) {
      const unsigned short* g =
          xc + ((((size_t)bi * HWDIM + gy) * HWDIM + gx) << 6) + (part << 3);
      unsigned ldsb = (unsigned)(((pix << 6) + (part << 3)) * 2);
      asm volatile("global_load_async_to_lds_b128 %0, %1, off"
                   :: "v"(ldsb), "v"((unsigned long long)(uintptr_t)g)
                   : "memory");
    }
  }
  asm volatile("s_wait_asynccnt 0" ::: "memory");
  __syncthreads();

  const int wave = tid >> 5;
  const int lane = tid & 31;
  const int m  = lane & 15;                // pixel-in-tile (A rows) / oc (C cols)
  const int hi = lane >> 4;

  // phase 1: the three dilated convs, WMMA per tap per K-chunk
  for (int mt = wave; mt < 16; mt += 8) {
    const int r  = mt >> 1;                // tile row 0..7
    const int cg = mt & 1;                 // 16-pixel column group
    for (int c = 0; c < 3; ++c) {
      const int dil = c + 1;
      const float* bias = (c == 0) ? b1 : (c == 1) ? b2 : b3;
      float bv = (m < 6) ? bias[m] : 0.f;
      v8f acc = {bv, bv, bv, bv, bv, bv, bv, bv};
      for (int ty = 0; ty < 5; ++ty) {
        for (int tx = 0; tx < 5; ++tx) {
          const int iy = r + HALO + (ty - 2) * dil;
          const int ix = cg * 16 + HALO + (tx - 2) * dil + m;
          const int pixbase = (iy * IW + ix) * 64;
          const int frag0 = ((c * 25 + (ty * 5 + tx)) * 2) * 512;
          __builtin_prefetch(pw + frag0 + 1024, 0, 0);   // next tap's fragments
#pragma unroll
          for (int kc = 0; kc < 2; ++kc) {
            const unsigned short* ap = xt + pixbase + kc * 32 + hi * 16;
            v8u a0 = *(const v8u*)ap;
            v8u a1 = *(const v8u*)(ap + 8);
            v16bf a = cat16(a0, a1);
            v16u bw = *(const v16u*)(pw + frag0 + kc * 512 + lane * 16);
            acc = __builtin_amdgcn_wmma_f32_16x16x32_bf16(
                false, a, false, __builtin_bit_cast(v16bf, bw),
                (short)0, acc, false, false);
          }
        }
      }
#pragma unroll
      for (int v = 0; v < 8; ++v) acc[v] = fmaxf(acc[v], 0.f);
      if (m < 6) {                         // valid output channels
        const int ch  = 36 + c * 6 + m;    // feat goes to out channels 36..53
        const int gy  = by * THT + r;
        const int gxb = bx * TW + cg * 16 + hi * 8;
        float* op = out + ((((size_t)bi * 55 + ch) * HWDIM + gy) * HWDIM + gxb);
        *(float4*)op       = make_float4(acc[0], acc[1], acc[2], acc[3]);
        *(float4*)(op + 4) = make_float4(acc[4], acc[5], acc[6], acc[7]);
        const int fch = c * 6 + m;
#pragma unroll
        for (int v = 0; v < 8; ++v) {
          int p = r * TW + cg * 16 + hi * 8 + v;
          ft[p * 32 + fch] = f2bf(acc[v]);
        }
      }
    }
  }
  __syncthreads();

  // phase 2: 1x1 convs (top 18->36, bot 18->1) + per-cell sum atomics
  for (int mt = wave; mt < 16; mt += 8) {
    const int r  = mt >> 1;
    const int cg = mt & 1;
    const int p  = r * TW + cg * 16 + m;
    const unsigned short* ap = ft + p * 32 + hi * 16;
    v8u a0 = *(const v8u*)ap;
    v8u a1 = *(const v8u*)(ap + 8);
    v16bf a = cat16(a0, a1);

    float s[8];
#pragma unroll
    for (int v = 0; v < 8; ++v) s[v] = 0.f;
#pragma unroll
    for (int nt = 0; nt < 3; ++nt) {
      int oc = nt * 16 + m;
      float bv = (oc < 36) ? bt[oc] : 0.f;
      v8f acc = {bv, bv, bv, bv, bv, bv, bv, bv};
      v16u bw = *(const v16u*)(pwt + nt * 512 + lane * 16);
      acc = __builtin_amdgcn_wmma_f32_16x16x32_bf16(
          false, a, false, __builtin_bit_cast(v16bf, bw),
          (short)0, acc, false, false);
#pragma unroll
      for (int v = 0; v < 8; ++v) s[v] += acc[v];
    }
    float bvb = (m == 0) ? bbv[0] : 0.f;
    v8f accB = {bvb, bvb, bvb, bvb, bvb, bvb, bvb, bvb};
    {
      v16u bw = *(const v16u*)(pwb + lane * 16);
      accB = __builtin_amdgcn_wmma_f32_16x16x32_bf16(
          false, a, false, __builtin_bit_cast(v16bf, bw),
          (short)0, accB, false, false);
    }
    // reduce top over the 16 output channels held across each 16-lane half
#pragma unroll
    for (int v = 0; v < 8; ++v) {
      float sv = s[v];
      sv += __shfl_xor(sv, 1, 32);
      sv += __shfl_xor(sv, 2, 32);
      sv += __shfl_xor(sv, 4, 32);
      sv += __shfl_xor(sv, 8, 32);
      s[v] = sv;
    }
    if (m == 0) {                          // lanes 0 (pixels 0..7) and 16 (8..15)
      const int gy  = by * THT + r;
      const int gxb = bx * TW + cg * 16 + hi * 8;
      const int rs  = rowseg[gy];
#pragma unroll
      for (int v = 0; v < 8; ++v) {
        const int cell = rs * NCG + colseg[gxb + v];
        atomicAdd(&topsum[cell], s[v]);
        atomicAdd(&botsum[cell], accB[v]);
      }
    }
  }
}

__global__ void finalize_cells(const int* __restrict__ rowseg, const int* __restrict__ colseg,
                               const float* __restrict__ ts, const float* __restrict__ bs,
                               float* __restrict__ tm, float* __restrict__ sg,
                               float* __restrict__ pools) {
  __shared__ int rc[NRG];
  __shared__ int cc[NCG];
  int t = threadIdx.x;
  if (t < NRG) { int n = 0; for (int i = 0; i < HWDIM; ++i) n += (rowseg[i] == t); rc[t] = n; }
  if (t < NCG) { int n = 0; for (int i = 0; i < HWDIM; ++i) n += (colseg[i] == t); cc[t] = n; }
  __syncthreads();
  if (t < NRG * NCG) {
    float cnt = (float)rc[t / NCG] * (float)cc[t % NCG];
    float tmv = ts[t] / (cnt * (float)(NB * 36));
    float bmv = bs[t] / (cnt * (float)NB);
    float sgv = 1.f / (1.f + __expf(-bmv));
    tm[t] = tmv;
    sg[t] = sgv;
    for (int b = 0; b < NB; ++b) pools[b * (NRG * NCG) + t] = sgv;
  }
}

__global__ __launch_bounds__(256) void broadcast_out(
    const int* __restrict__ rowseg, const int* __restrict__ colseg,
    const float* __restrict__ tm, const float* __restrict__ sg,
    float* __restrict__ out) {
  int idx = blockIdx.x * 256 + threadIdx.x;   // H*W pixels
  int x = idx & (HWDIM - 1);
  int y = idx >> 9;
  int cell = rowseg[y] * NCG + colseg[x];
  float tmv = tm[cell], sgv = sg[cell];
  size_t pix = (size_t)y * HWDIM + x;
  for (int b = 0; b < NB; ++b) {
    size_t base = (size_t)b * 55 * HWDIM * HWDIM + pix;
#pragma unroll
    for (int ch = 0; ch < 36; ++ch)
      out[base + (size_t)ch * HWDIM * HWDIM] = tmv;
    out[base + (size_t)54 * HWDIM * HWDIM] = sgv;
  }
}

// ---------------- launch ----------------
extern "C" void kernel_launch(void* const* d_in, const int* in_sizes, int n_in,
                              void* d_out, int out_size, void* d_ws, size_t ws_size,
                              hipStream_t stream) {
  const float* x      = (const float*)d_in[0];
  const int*   rowseg = (const int*)d_in[1];
  const int*   colseg = (const int*)d_in[2];
  const float* w1 = (const float*)d_in[3];
  const float* b1 = (const float*)d_in[4];
  const float* w2 = (const float*)d_in[5];
  const float* b2 = (const float*)d_in[6];
  const float* w3 = (const float*)d_in[7];
  const float* b3 = (const float*)d_in[8];
  const float* wt = (const float*)d_in[9];
  const float* bt = (const float*)d_in[10];
  const float* wb = (const float*)d_in[11];
  const float* bbv = (const float*)d_in[12];
  float* out = (float*)d_out;
  char*  ws  = (char*)d_ws;

  unsigned short* xc  = (unsigned short*)(ws + OFF_XC);
  unsigned short* pw  = (unsigned short*)(ws + OFF_PW);
  unsigned short* pwt = (unsigned short*)(ws + OFF_PWT);
  unsigned short* pwb = (unsigned short*)(ws + OFF_PWB);
  float* ts = (float*)(ws + OFF_TS);
  float* bs = (float*)(ws + OFF_BS);
  float* tm = (float*)(ws + OFF_TM);
  float* sg = (float*)(ws + OFF_SG);

  zero_accum<<<1, 1024, 0, stream>>>(ts, bs);
  cast_x<<<(NB * HWDIM * HWDIM) / 256, 256, 0, stream>>>(x, xc);
  pack_conv_w<<<150, 32, 0, stream>>>(w1, w2, w3, pw);
  pack_wt<<<3, 32, 0, stream>>>(wt, pwt);
  pack_wb<<<1, 32, 0, stream>>>(wb, pwb);

  dim3 grid(HWDIM / TW, HWDIM / THT, NB);   // 16 x 64 x 4
  size_t smem = (size_t)(XT_HALFS + FT_HALFS) * 2;  // 129,024 B (< 320 KB/WGP)
  conv_main<<<grid, 256, smem, stream>>>(xc, pw, pwt, pwb, b1, b2, b3, bt, bbv,
                                         rowseg, colseg, out, ts, bs);

  float* pools = out + (size_t)NB * 55 * HWDIM * HWDIM;
  finalize_cells<<<1, 640, 0, stream>>>(rowseg, colseg, ts, bs, tm, sg, pools);
  broadcast_out<<<(HWDIM * HWDIM) / 256, 256, 0, stream>>>(rowseg, colseg, tm, sg, out);
}
